// MultiHeadSelfAttention_56727928045932
// MI455X (gfx1250) — compile-verified, matrix operand order checked by reading, not run on hardware
//
#include <hip/hip_runtime.h>

#define DEV __device__ __forceinline__

typedef __bf16 bf16_t;
typedef __attribute__((ext_vector_type(16))) __bf16 v16bf;
typedef __attribute__((ext_vector_type(8)))  __bf16 v8bf;
typedef __attribute__((ext_vector_type(4)))  __bf16 v4bf;
typedef __attribute__((ext_vector_type(8)))  float   v8f;
typedef __attribute__((ext_vector_type(4)))  float   v4f;

// Problem constants (from reference): B=2, S=2048, D=1024, H=16, HD=64
static constexpr int Bn  = 2;
static constexpr int Sn  = 2048;
static constexpr int Dn  = 1024;
static constexpr int Hn  = 16;
static constexpr int HDn = 64;
static constexpr int Mn  = Bn * Sn;   // 4096 rows for all projection GEMMs

// ---------- helpers ----------

// f32 -> bf16 with round-to-nearest-even (bit manipulation; no reliance on
// __bf16 conversion semantics).
DEV bf16_t f2bf(float f) {
  union { float f; unsigned u; } v; v.f = f;
  unsigned r = v.u + 0x7FFFu + ((v.u >> 16) & 1u);
  unsigned short h = (unsigned short)(r >> 16);
  union { unsigned short s; bf16_t b; } o; o.s = h;
  return o.b;
}

DEV v8f wmma_bf16(v16bf a, v16bf b, v8f c) {
  // D = A(16x32 bf16) * B(32x16 bf16) + C(16x16 f32)
  return __builtin_amdgcn_wmma_f32_16x16x32_bf16(
      /*neg_a=*/false, a, /*neg_b=*/false, b,
      /*c_mod=*/(short)0, c, /*reuse_a=*/false, /*reuse_b=*/false);
}

// 32-bit LDS byte offset for an LDS object: flat->LDS mapping keeps the
// offset in addr[31:0] (ISA 10.2 aperture rules), so truncation is exact.
DEV unsigned lds_addr(const void* p) { return (unsigned)(size_t)p; }

// gfx1250 async direct global->LDS copy (no VGPR staging, ASYNCcnt-tracked).
DEV void async_ld_b128(unsigned lds_off, const void* gaddr) {
  asm volatile("global_load_async_to_lds_b128 %0, %1, off"
               :: "v"(lds_off), "v"(gaddr) : "memory");
}
DEV void wait_async0() {
  asm volatile("s_wait_asynccnt 0x0" ::: "memory");
}

// Load a 16x32 bf16 operand fragment from an LDS tile stored row-major with
// `stride` elements per row. Per the CDNA5 ISA A-matrix layout (16-bit,
// 16x32): lane L holds row M=L%16; lanes 0-15 hold K={0..7,16..23},
// lanes 16-31 hold K={8..15,24..31}; v16 elements 0..7 = low K-half,
// 8..15 = high K-half. Two ds_load_b128 per fragment.
// B-operand uses the mirrored convention (lane = column) so storing the
// B matrix K-contiguous (i.e. as B^T rows) lets us use the same loader.
DEV v16bf load_frag(const bf16_t* base, int stride) {
  const int lane = (int)(threadIdx.x & 31);
  const int r    = lane & 15;
  const int ko   = (lane >> 4) << 3;   // 0 or 8
  const bf16_t* p = base + r * stride;
  v8bf lo = *(const v8bf*)(p + ko);        // K = ko .. ko+7
  v8bf hi = *(const v8bf*)(p + 16 + ko);   // K = 16+ko .. 23+ko
  v16bf f;
#pragma unroll
  for (int i = 0; i < 8; ++i) { f[i] = lo[i]; f[i + 8] = hi[i]; }
  return f;
}

// ---------- precision-conversion kernels (one-time, bandwidth-trivial) ----------

__global__ __launch_bounds__(256) void convert_x_kernel(const float* __restrict__ in,
                                                        bf16_t* __restrict__ out) {
  size_t i = ((size_t)blockIdx.x * 256 + threadIdx.x) * 4;
  v4f v = *(const v4f*)(in + i);
  v4bf o;
#pragma unroll
  for (int j = 0; j < 4; ++j) o[j] = f2bf(v[j]);
  *(v4bf*)(out + i) = o;
}

// W [K=1024][N=1024] f32  ->  Wt [N][K] bf16 (transposed so the GEMM's
// B operand is K-contiguous).
__global__ __launch_bounds__(256) void transpose_w_kernel(const float* __restrict__ W,
                                                          bf16_t* __restrict__ Wt) {
  int t  = blockIdx.x * 256 + threadIdx.x;   // 0 .. 262143
  int n  = t & (Dn - 1);
  int k  = (t >> 10) * 4;
  v4bf o;
#pragma unroll
  for (int i = 0; i < 4; ++i) o[i] = f2bf(W[(size_t)(k + i) * Dn + n]);
  *(v4bf*)(Wt + (size_t)n * Dn + k) = o;
}

// ---------- WMMA GEMM: C[M=4096,N=1024] = A[M,K] * Bt[N,K]^T + bias ----------

enum { OUT_QK = 0, OUT_VT = 1, OUT_F32 = 2 };

template <int MODE>
__global__ __launch_bounds__(256)
void gemm128_kernel(const bf16_t* __restrict__ A,    // [M, K] row-major bf16
                    const bf16_t* __restrict__ Bt,   // [N, K] row-major bf16
                    const float*  __restrict__ bias, // [N]
                    void* __restrict__ Cout) {
  constexpr int Kdim = Dn;           // 1024
  constexpr int LDT  = 40;           // padded LDS stride (elements): spreads banks, 16B aligned
  __shared__ bf16_t sA[2][128 * LDT];
  __shared__ bf16_t sB[2][128 * LDT];

  const int tid   = threadIdx.x;
  const int wave  = tid >> 5;
  const int lane  = tid & 31;
  const int mBase = blockIdx.x * 128;            // grid.x = 32
  const int nBase = blockIdx.y * 128;            // grid.y = 8
  const int wm    = (wave >> 2) * 64;            // wave 2x4 grid: 64x32 per wave
  const int wn    = (wave & 3) * 32;

  // Global->LDS async staging: 256 threads x 4 chunks of 8 bf16 per k-step.
  const int row0 = tid >> 2;          // 0..63
  const int kc0  = (tid & 3) * 8;     // 0,8,16,24

  auto issueTiles = [&](int kBase, int buf) {
    const bf16_t* ga = A  + (size_t)mBase * Kdim + kBase;
    const bf16_t* gb = Bt + (size_t)nBase * Kdim + kBase;
    async_ld_b128(lds_addr(&sA[buf][row0        * LDT + kc0]), ga + (size_t)row0        * Kdim + kc0);
    async_ld_b128(lds_addr(&sA[buf][(row0 + 64) * LDT + kc0]), ga + (size_t)(row0 + 64) * Kdim + kc0);
    async_ld_b128(lds_addr(&sB[buf][row0        * LDT + kc0]), gb + (size_t)row0        * Kdim + kc0);
    async_ld_b128(lds_addr(&sB[buf][(row0 + 64) * LDT + kc0]), gb + (size_t)(row0 + 64) * Kdim + kc0);
  };

  v8f acc[4][2];
#pragma unroll
  for (int mt = 0; mt < 4; ++mt)
#pragma unroll
    for (int nt = 0; nt < 2; ++nt) acc[mt][nt] = (v8f)(0.0f);

  issueTiles(0, 0);

  constexpr int NKT = Kdim / 32;   // 32 K-steps
  for (int kt = 0; kt < NKT; ++kt) {
    const int cur = kt & 1;
    wait_async0();                 // my async writes into buf[cur] landed
    __syncthreads();               // everyone's landed; prev reads of buf[1-cur] done
    if (kt + 1 < NKT) issueTiles((kt + 1) * 32, 1 - cur);
    if (kt + 2 < NKT) {            // keep L2/L0 warm two tiles ahead
      __builtin_prefetch(A  + (size_t)(mBase + row0) * Kdim + (kt + 2) * 32 + kc0, 0, 1);
      __builtin_prefetch(Bt + (size_t)(nBase + row0) * Kdim + (kt + 2) * 32 + kc0, 0, 1);
    }

    v16bf aF[4], bF[2];
#pragma unroll
    for (int mt = 0; mt < 4; ++mt) aF[mt] = load_frag(&sA[cur][(wm + mt * 16) * LDT], LDT);
#pragma unroll
    for (int nt = 0; nt < 2; ++nt) bF[nt] = load_frag(&sB[cur][(wn + nt * 16) * LDT], LDT);
#pragma unroll
    for (int mt = 0; mt < 4; ++mt)
#pragma unroll
      for (int nt = 0; nt < 2; ++nt)
        acc[mt][nt] = wmma_bf16(aF[mt], bF[nt], acc[mt][nt]);
  }

  // Epilogue. C layout: vgpr i, lanes 0-15 -> row i, col lane; lanes 16-31 -> row i+8.
  const int half8 = (lane >> 4) << 3;
#pragma unroll
  for (int mt = 0; mt < 4; ++mt)
#pragma unroll
    for (int nt = 0; nt < 2; ++nt)
#pragma unroll
      for (int i = 0; i < 8; ++i) {
        const int r = mBase + wm + mt * 16 + i + half8;
        const int c = nBase + wn + nt * 16 + (lane & 15);
        const float v = acc[mt][nt][i] + bias[c];
        if (MODE == OUT_F32) {
          ((float*)Cout)[(size_t)r * Dn + c] = v;
        } else {
          const int b = r >> 11, s = r & (Sn - 1);
          const int h = c >> 6,  d = c & (HDn - 1);
          if (MODE == OUT_QK)   // [B,H,S,HD] bf16
            ((bf16_t*)Cout)[(((size_t)b * Hn + h) * Sn + s) * HDn + d] = f2bf(v);
          else                  // OUT_VT: [B,H,HD,S] bf16 (V transposed for PV WMMA)
            ((bf16_t*)Cout)[(((size_t)b * Hn + h) * HDn + d) * Sn + s] = f2bf(v);
        }
      }
}

// ---------- flash attention: per block one (b,h,128-row q-block) ----------

__global__ __launch_bounds__(256)
void flash_attn_kernel(const bf16_t* __restrict__ Q,   // [B,H,S,HD]
                       const bf16_t* __restrict__ Km,  // [B,H,S,HD]
                       const bf16_t* __restrict__ Vt,  // [B,H,HD,S]
                       bf16_t* __restrict__ O) {       // [B,S,D] (head h -> cols h*64..)
  constexpr int LQK = 72;   // padded stride for 64-wide bf16 tiles (144B, 16B-aligned)
  __shared__ bf16_t sQ[128 * LQK];   // 18.0 KB
  __shared__ bf16_t sK[ 64 * LQK];   //  9.0 KB
  __shared__ bf16_t sV[ 64 * LQK];   //  9.0 KB  (Vt tile: rows=d, cols=s)
  __shared__ bf16_t sP[128 * LQK];   // 18.0 KB  (per-wave private 16-row slices)

  const int tid  = threadIdx.x;
  const int lane = tid & 31;
  const int wave = tid >> 5;
  const int bh   = blockIdx.x >> 4;          // grid.x = B*H*(S/128) = 512
  const int qb   = blockIdx.x & 15;
  const int b    = bh >> 4;                  // H = 16
  const int h    = bh & 15;
  const size_t qkBase = (size_t)bh * Sn * HDn;
  const size_t vBase  = (size_t)bh * HDn * Sn;

  // Stage the 128x64 Q tile once (async direct-to-LDS).
  {
    const bf16_t* g = Q + qkBase + (size_t)qb * 128 * HDn;
#pragma unroll
    for (int t = tid; t < 128 * HDn / 8; t += 256) {
      const int r = t >> 3, kc = (t & 7) * 8;
      async_ld_b128(lds_addr(&sQ[r * LQK + kc]), g + (size_t)r * HDn + kc);
    }
  }
  wait_async0();
  __syncthreads();

  const int wq = wave * 16;                  // this wave's 16 q-rows
  const v16bf aQ0 = load_frag(&sQ[wq * LQK],      LQK);   // K = 0..31
  const v16bf aQ1 = load_frag(&sQ[wq * LQK + 32], LQK);   // K = 32..63

  v8f accO[4];
#pragma unroll
  for (int dt = 0; dt < 4; ++dt) accO[dt] = (v8f)(0.0f);
  float mrun[8], lrun[8];
#pragma unroll
  for (int i = 0; i < 8; ++i) { mrun[i] = -1e30f; lrun[i] = 0.0f; }

  const float scale = 0.125f;                // 1/sqrt(64)
  const int half8 = (lane >> 4) << 3;

  for (int kb = 0; kb < Sn / 64; ++kb) {     // 32 key blocks of 64
    __syncthreads();                         // prev iter's sK/sV reads complete
    {
      const bf16_t* gk = Km + qkBase + (size_t)kb * 64 * HDn;
      const bf16_t* gv = Vt + vBase + (size_t)kb * 64;
#pragma unroll
      for (int t = tid; t < 64 * HDn / 8; t += 256) {      // 64x64 K tile
        const int r = t >> 3, kc = (t & 7) * 8;
        async_ld_b128(lds_addr(&sK[r * LQK + kc]), gk + (size_t)r * HDn + kc);
      }
#pragma unroll
      for (int t = tid; t < HDn * 64 / 8; t += 256) {      // 64(d) x 64(s) V^T tile
        const int d = t >> 3, sc = (t & 7) * 8;
        async_ld_b128(lds_addr(&sV[d * LQK + sc]), gv + (size_t)d * Sn + sc);
      }
    }
    wait_async0();
    __syncthreads();

    // S = Q * K^T  (4 column tiles of 16; K-dim = 64 -> 2 WMMA each)
    float sc[4][8];
#pragma unroll
    for (int nt = 0; nt < 4; ++nt) {
      const v16bf b0 = load_frag(&sK[(nt * 16) * LQK],      LQK);
      const v16bf b1 = load_frag(&sK[(nt * 16) * LQK + 32], LQK);
      v8f c = (v8f)(0.0f);
      c = wmma_bf16(aQ0, b0, c);
      c = wmma_bf16(aQ1, b1, c);
#pragma unroll
      for (int i = 0; i < 8; ++i) sc[nt][i] = c[i] * scale;
    }

    // Online softmax. Row (i + 8*half) lives in one 16-lane half of vgpr i.
#pragma unroll
    for (int i = 0; i < 8; ++i) {
      float v = -1e30f;
#pragma unroll
      for (int nt = 0; nt < 4; ++nt) v = fmaxf(v, sc[nt][i]);
#pragma unroll
      for (int off = 8; off >= 1; off >>= 1) v = fmaxf(v, __shfl_xor(v, off, 32));
      const float mn   = fmaxf(mrun[i], v);
      const float corr = __expf(mrun[i] - mn);
      mrun[i] = mn;
      lrun[i] *= corr;
#pragma unroll
      for (int dt = 0; dt < 4; ++dt) accO[dt][i] *= corr;

      float ps = 0.0f;
#pragma unroll
      for (int nt = 0; nt < 4; ++nt) {
        const float p = __expf(sc[nt][i] - mn);
        ps += p;
        sP[(wq + i + half8) * LQK + nt * 16 + (lane & 15)] = f2bf(p);
      }
#pragma unroll
      for (int off = 8; off >= 1; off >>= 1) ps += __shfl_xor(ps, off, 32);
      lrun[i] += ps;
    }

    // O += P(16x64) * V(64x64): per-wave-private sP region, no barrier needed.
    v16bf aP[2];
#pragma unroll
    for (int kc = 0; kc < 2; ++kc) aP[kc] = load_frag(&sP[wq * LQK + kc * 32], LQK);
#pragma unroll
    for (int dt = 0; dt < 4; ++dt) {
      v8f c = accO[dt];
#pragma unroll
      for (int kc = 0; kc < 2; ++kc) {
        const v16bf bV = load_frag(&sV[(dt * 16) * LQK + kc * 32], LQK);
        c = wmma_bf16(aP[kc], bV, c);
      }
      accO[dt] = c;
    }
  }

  // Write normalized output as bf16 rows of the [B,S,D] pre-projection matrix.
#pragma unroll
  for (int dt = 0; dt < 4; ++dt)
#pragma unroll
    for (int i = 0; i < 8; ++i) {
      const int s   = qb * 128 + wq + i + half8;
      const int col = h * HDn + dt * 16 + (lane & 15);
      const float inv = 1.0f / lrun[i];
      O[((size_t)b * Sn + s) * Dn + col] = f2bf(accO[dt][i] * inv);
    }
}

// ---------- host launcher ----------

extern "C" void kernel_launch(void* const* d_in, const int* in_sizes, int n_in,
                              void* d_out, int out_size, void* d_ws, size_t ws_size,
                              hipStream_t stream) {
  (void)in_sizes; (void)n_in; (void)out_size; (void)ws_size;
  const float* x  = (const float*)d_in[0];
  const float* Wq = (const float*)d_in[1];
  const float* bq = (const float*)d_in[2];
  const float* Wk = (const float*)d_in[3];
  const float* bk = (const float*)d_in[4];
  const float* Wv = (const float*)d_in[5];
  const float* bv = (const float*)d_in[6];
  const float* Wo = (const float*)d_in[7];
  const float* bo = (const float*)d_in[8];
  float* out = (float*)d_out;

  // Workspace carve-up (bf16 everywhere): ~50.3 MB total.
  char* ws = (char*)d_ws;
  size_t off = 0;
  auto carve = [&](size_t elems) {
    bf16_t* p = (bf16_t*)(ws + off);
    off = (off + elems * sizeof(bf16_t) + 255) & ~(size_t)255;
    return p;
  };
  bf16_t* xb  = carve((size_t)Mn * Dn);
  bf16_t* wqT = carve((size_t)Dn * Dn);
  bf16_t* wkT = carve((size_t)Dn * Dn);
  bf16_t* wvT = carve((size_t)Dn * Dn);
  bf16_t* woT = carve((size_t)Dn * Dn);
  bf16_t* Qb  = carve((size_t)Mn * Dn);
  bf16_t* Kb  = carve((size_t)Mn * Dn);
  bf16_t* Vtb = carve((size_t)Mn * Dn);
  bf16_t* Ab  = carve((size_t)Mn * Dn);

  convert_x_kernel<<<(Mn * Dn) / (4 * 256), 256, 0, stream>>>(x, xb);
  transpose_w_kernel<<<(Dn * Dn) / (4 * 256), 256, 0, stream>>>(Wq, wqT);
  transpose_w_kernel<<<(Dn * Dn) / (4 * 256), 256, 0, stream>>>(Wk, wkT);
  transpose_w_kernel<<<(Dn * Dn) / (4 * 256), 256, 0, stream>>>(Wv, wvT);
  transpose_w_kernel<<<(Dn * Dn) / (4 * 256), 256, 0, stream>>>(Wo, woT);

  dim3 gg(Mn / 128, Dn / 128);  // 32 x 8
  gemm128_kernel<OUT_QK><<<gg, 256, 0, stream>>>(xb, wqT, bq, (void*)Qb);
  gemm128_kernel<OUT_QK><<<gg, 256, 0, stream>>>(xb, wkT, bk, (void*)Kb);
  gemm128_kernel<OUT_VT><<<gg, 256, 0, stream>>>(xb, wvT, bv, (void*)Vtb);

  flash_attn_kernel<<<Bn * Hn * (Sn / 128), 256, 0, stream>>>(Qb, Kb, Vtb, Ab);

  gemm128_kernel<OUT_F32><<<gg, 256, 0, stream>>>(Ab, woT, bo, (void*)out);
}
